// PointEncoder_80556406603869
// MI455X (gfx1250) — matrix operand order repaired and checked by reading, not compiled, not run on hardware
//
#include <hip/hip_runtime.h>
#include <hip/hip_bf16.h>

#define B_ 32
#define N_ 16384
#define D_ 128
#define H_ 64
#define S_ 256

typedef _Float16 v16h __attribute__((ext_vector_type(16)));
typedef _Float16 v8h  __attribute__((ext_vector_type(8)));
typedef float    v8f  __attribute__((ext_vector_type(8)));

__device__ __forceinline__ float gelu_exact(float x) {
    return 0.5f * x * (1.0f + erff(x * 0.70710678118654752440f));
}

// ---- WMMA fragment helpers (gfx1250 wave32 layouts, ISA 7.12.2) ----
// A (16x32 f16): lane L holds row M=L&15; halves e=0..7 -> K = kk*32 + (L<16?0:8) + e,
//                halves e=8..15 -> same + 16.  Two contiguous 16-byte chunks.
__device__ __forceinline__ v16h ld_frag_2x8(const _Float16* p0, const _Float16* p1) {
    v8h lo = *(const v8h*)p0;
    v8h hi = *(const v8h*)p1;
    v16h r;
#pragma unroll
    for (int e = 0; e < 8; ++e) { r[e] = lo[e]; r[e + 8] = hi[e]; }
    return r;
}

__device__ __forceinline__ v16h load_a_frag(const _Float16* base, int lane, int kk, int ld) {
    const int m  = lane & 15;
    const int k0 = kk * 32 + ((lane & 16) ? 8 : 0);
    const _Float16* row = base + m * ld + k0;
    return ld_frag_2x8(row, row + 16);
}

// B (32x16 f16) from transposed weights wT[n][k]: lane L holds column N=col0+(L&15),
// halves e=0..7 -> K = kb + (L<16?0:8) + e, halves e=8..15 -> same + 16.
__device__ __forceinline__ v16h load_b_frag(const _Float16* wT, int lane, int col0, int kb, int ld) {
    const int n  = col0 + (lane & 15);
    const int k0 = kb + ((lane & 16) ? 8 : 0);
    const _Float16* row = wT + n * ld + k0;
    return ld_frag_2x8(row, row + 16);
}

__device__ __forceinline__ v8f wmma32(v16h a, v16h b, v8f c) {
    return __builtin_amdgcn_wmma_f32_16x16x32_f16(false, a, false, b, (short)0, c, false, false);
}

// ---------------- zero workspace ----------------
__global__ void k_zero(float* p, int n) {
    int i = blockIdx.x * blockDim.x + threadIdx.x;
    if (i < n) p[i] = 0.0f;
}

// ---------------- kernel 1: per-point MLP 3->64->128 + mean accumulation ----------------
// 256 threads = 8 waves; each wave owns 16 points; WG covers 128 points.
__global__ __launch_bounds__(256) void k_feats(
    const float* __restrict__ points, const float* __restrict__ W1, const float* __restrict__ b1,
    const float* __restrict__ W2, const float* __restrict__ b2,
    _Float16* __restrict__ feats, float* __restrict__ gsum)
{
    __shared__ _Float16 sW2T[D_ * H_];   // [n][k]  16KB
    __shared__ _Float16 sH[8][16 * H_];  // per-wave hidden, 16KB
    __shared__ float    sW1[3 * H_];
    __shared__ float    sb1[H_];
    __shared__ float    sb2[D_];
    __shared__ float    sP[128 * 3];

    const int tid  = threadIdx.x;
    const int wave = tid >> 5;
    const int lane = tid & 31;
    const int ptBase = blockIdx.x * 128;          // global row across B*N
    const int b = ptBase / N_;

    // stage weights / biases / points
    for (int i = tid; i < D_ * H_; i += 256) {
        int n = i >> 6, k = i & 63;
        sW2T[n * H_ + k] = (_Float16)W2[k * D_ + n];
    }
    for (int i = tid; i < 3 * H_; i += 256) sW1[i] = W1[i];
    if (tid < H_) sb1[tid] = b1[tid];
    if (tid < D_) sb2[tid] = b2[tid];
    for (int i = tid; i < 128 * 3; i += 256) sP[i] = points[(size_t)ptBase * 3 + i];
    __syncthreads();

    // layer 1 (K=3) scalar: each wave fills its 16x64 hidden tile
    const int row0 = wave * 16;
    _Float16* Hl = sH[wave];
#pragma unroll
    for (int i = 0; i < 32; ++i) {
        int idx = i * 32 + lane;                  // 0..1023
        int r = idx >> 6, c = idx & 63;
        float x = sP[(row0 + r) * 3 + 0];
        float y = sP[(row0 + r) * 3 + 1];
        float z = sP[(row0 + r) * 3 + 2];
        float h = x * sW1[c] + y * sW1[H_ + c] + z * sW1[2 * H_ + c] + sb1[c];
        Hl[r * H_ + c] = (_Float16)gelu_exact(h);
    }
    __syncthreads();

    // layer 2 via WMMA: [16x64] @ [64x128], K-steps of 32, 8 N-tiles
    v16h a0 = load_a_frag(Hl, lane, 0, H_);
    v16h a1 = load_a_frag(Hl, lane, 1, H_);
    const int gRow0 = ptBase + row0;
    const int nLane = lane & 15;
    const int mOff  = (lane & 16) ? 8 : 0;
#pragma unroll
    for (int nt = 0; nt < 8; ++nt) {
        const int col0 = nt * 16;
        const int nCol = col0 + nLane;
        float bias = sb2[nCol];
        v8f c = {bias, bias, bias, bias, bias, bias, bias, bias};
        c = wmma32(a0, load_b_frag(sW2T, lane, col0, 0,  H_), c);
        c = wmma32(a1, load_b_frag(sW2T, lane, col0, 32, H_), c);
        float psum = 0.0f;
#pragma unroll
        for (int r = 0; r < 8; ++r) {
            float v = gelu_exact(c[r]);
            feats[(size_t)(gRow0 + mOff + r) * D_ + nCol] = (_Float16)v;
            psum += v;
        }
        psum += __shfl_down(psum, 16, 32);        // fold rows 8..15 into 0..7 half-wave
        if (lane < 16) atomicAdd(&gsum[b * D_ + nCol], psum);
    }
}

// ---------------- kernel 2: farthest point sampling (1 WG per batch) ----------------
__global__ __launch_bounds__(256) void k_fps(const float* __restrict__ points,
                                             int* __restrict__ idxOut)
{
    __shared__ float md[N_];                      // 64 KB min-distance table
    __shared__ float sVal[8];
    __shared__ int   sIdx[8];
    __shared__ int   sFar;

    const int tid  = threadIdx.x;
    const int lane = tid & 31;
    const int wave = tid >> 5;
    const int b    = blockIdx.x;
    const float* base = points + (size_t)b * N_ * 3;

    for (int i = tid; i < N_; i += 256) md[i] = 3.4e38f;
    if (tid == 0) sFar = 0;
    __syncthreads();

    for (int s = 0; s < S_; ++s) {
        const int far = sFar;
        if (tid == 0) idxOut[b * S_ + s] = far;
        const float sx = base[far * 3 + 0];
        const float sy = base[far * 3 + 1];
        const float sz = base[far * 3 + 2];

        float bv = -1.0f; int bi = N_;
        for (int i = tid; i < N_; i += 256) {
            float dx = base[i * 3 + 0] - sx;
            float dy = base[i * 3 + 1] - sy;
            float dz = base[i * 3 + 2] - sz;
            float d = dx * dx + dy * dy + dz * dz;
            float m = md[i];
            m = fminf(m, d);
            md[i] = m;
            if (m > bv) { bv = m; bi = i; }
        }
        // wave32 argmax reduce (ties -> lowest index, matching argmax-first)
#pragma unroll
        for (int o = 16; o > 0; o >>= 1) {
            float ov = __shfl_down(bv, (unsigned)o, 32);
            int   oi = __shfl_down(bi, (unsigned)o, 32);
            if (ov > bv || (ov == bv && oi < bi)) { bv = ov; bi = oi; }
        }
        if (lane == 0) { sVal[wave] = bv; sIdx[wave] = bi; }
        __syncthreads();
        if (tid == 0) {
            float best = sVal[0]; int bidx = sIdx[0];
#pragma unroll
            for (int w = 1; w < 8; ++w)
                if (sVal[w] > best || (sVal[w] == best && sIdx[w] < bidx)) { best = sVal[w]; bidx = sIdx[w]; }
            sFar = bidx;
        }
        __syncthreads();
    }
}

// ---------------- kernel 3: gather tokens, fused MLP 256->128->128 (WMMA), LayerNorm ----------------
// 128 threads = 4 waves; one WG per (batch, 16-token tile). Dynamic LDS ~116 KB.
__global__ __launch_bounds__(128) void k_tokens(
    const _Float16* __restrict__ feats, const float* __restrict__ gsum,
    const int* __restrict__ idx,
    const float* __restrict__ Wf1, const float* __restrict__ bf1,
    const float* __restrict__ Wf2, const float* __restrict__ bf2,
    const float* __restrict__ gamma, const float* __restrict__ beta,
    float* __restrict__ out)
{
    extern __shared__ char smem[];
    _Float16* sW1T = (_Float16*)smem;             // [128][256]  64 KB
    _Float16* sW2T = sW1T + 128 * 256;            // [128][128]  32 KB
    _Float16* sCat = sW2T + 128 * 128;            // [16][256]    8 KB
    _Float16* sU   = sCat + 16 * 256;             // [16][128]    4 KB
    float*    sO   = (float*)(sU + 16 * 128);     // [16][128]    8 KB
    float*    sSt  = sO + 16 * 128;               // [16][2]

    const int tid  = threadIdx.x;                 // 0..127
    const int wave = tid >> 5;
    const int lane = tid & 31;
    const int b    = blockIdx.x >> 4;
    const int tile = blockIdx.x & 15;
    const int tok0 = tile * 16;

    // stage transposed weights in f16
    for (int i = tid; i < 128 * 256; i += 128) { int n = i >> 8, k = i & 255; sW1T[n * 256 + k] = (_Float16)Wf1[k * D_ + n]; }
    for (int i = tid; i < 128 * 128; i += 128) { int n = i >> 7, k = i & 127; sW2T[n * 128 + k] = (_Float16)Wf2[k * D_ + n]; }

    // gather cat = [feats[idx], mean(feats)]
    const float invN = 1.0f / (float)N_;
    for (int i = tid; i < 16 * 256; i += 128) {
        int m = i >> 8, k = i & 255;
        int pt = idx[b * S_ + tok0 + m];
        _Float16 v;
        if (k < D_) v = feats[((size_t)b * N_ + pt) * D_ + k];
        else        v = (_Float16)(gsum[b * D_ + (k - D_)] * invN);
        sCat[m * 256 + k] = v;
    }
    __syncthreads();

    const int nLane = lane & 15;
    const int mOff  = (lane & 16) ? 8 : 0;

    // layer f1: [16x256] @ [256x128], 8 K-steps, 2 N-tiles per wave
#pragma unroll
    for (int t = 0; t < 2; ++t) {
        const int col0 = (wave * 2 + t) * 16;
        const int nCol = col0 + nLane;
        float bias = bf1[nCol];
        v8f c = {bias, bias, bias, bias, bias, bias, bias, bias};
#pragma unroll
        for (int kk = 0; kk < 8; ++kk)
            c = wmma32(load_a_frag(sCat, lane, kk, 256),
                       load_b_frag(sW1T, lane, col0, kk * 32, 256), c);
#pragma unroll
        for (int r = 0; r < 8; ++r)
            sU[(mOff + r) * D_ + nCol] = (_Float16)gelu_exact(c[r]);
    }
    __syncthreads();

    // layer f2: [16x128] @ [128x128], 4 K-steps
#pragma unroll
    for (int t = 0; t < 2; ++t) {
        const int col0 = (wave * 2 + t) * 16;
        const int nCol = col0 + nLane;
        float bias = bf2[nCol];
        v8f c = {bias, bias, bias, bias, bias, bias, bias, bias};
#pragma unroll
        for (int kk = 0; kk < 4; ++kk)
            c = wmma32(load_a_frag(sU, lane, kk, 128),
                       load_b_frag(sW2T, lane, col0, kk * 32, 128), c);
#pragma unroll
        for (int r = 0; r < 8; ++r)
            sO[(mOff + r) * D_ + nCol] = c[r];
    }
    __syncthreads();

    // LayerNorm over D=128 per token
    if (tid < 16) {
        float s = 0.0f, s2 = 0.0f;
        for (int k = 0; k < D_; ++k) { float v = sO[tid * D_ + k]; s += v; s2 += v * v; }
        float mu  = s * (1.0f / D_);
        float var = s2 * (1.0f / D_) - mu * mu;
        sSt[tid * 2 + 0] = mu;
        sSt[tid * 2 + 1] = rsqrtf(var + 1e-5f);
    }
    __syncthreads();
    for (int i = tid; i < 16 * D_; i += 128) {
        int m = i >> 7, k = i & 127;
        float v = (sO[m * D_ + k] - sSt[m * 2]) * sSt[m * 2 + 1] * gamma[k] + beta[k];
        out[((size_t)b * S_ + tok0 + m) * D_ + k] = v;
    }
}

extern "C" void kernel_launch(void* const* d_in, const int* in_sizes, int n_in,
                              void* d_out, int out_size, void* d_ws, size_t ws_size,
                              hipStream_t stream) {
    const float* points = (const float*)d_in[0];
    const float* W1     = (const float*)d_in[1];
    const float* b1     = (const float*)d_in[2];
    const float* W2     = (const float*)d_in[3];
    const float* b2     = (const float*)d_in[4];
    const float* Wf1    = (const float*)d_in[5];
    const float* bf1    = (const float*)d_in[6];
    const float* Wf2    = (const float*)d_in[7];
    const float* bf2    = (const float*)d_in[8];
    const float* gamma  = (const float*)d_in[9];
    const float* beta   = (const float*)d_in[10];
    float* out = (float*)d_out;

    // workspace layout
    char* ws = (char*)d_ws;
    _Float16* feats = (_Float16*)ws;                                  // B*N*D f16 = 128 MB
    float*    gsum  = (float*)(ws + (size_t)B_ * N_ * D_ * 2);        // B*D f32
    int*      idx   = (int*)((char*)gsum + (size_t)B_ * D_ * 4);      // B*S i32

    k_zero<<<(B_ * D_ + 255) / 256, 256, 0, stream>>>(gsum, B_ * D_);
    k_feats<<<(B_ * N_) / 128, 256, 0, stream>>>(points, W1, b1, W2, b2, feats, gsum);
    k_fps<<<B_, 256, 0, stream>>>(points, idx);

    const size_t smemBytes = (size_t)(128 * 256 + 128 * 128 + 16 * 256 + 16 * 128) * 2
                           + (size_t)(16 * 128 + 32) * 4;             // ~116 KB
    k_tokens<<<B_ * 16, 128, smemBytes, stream>>>(feats, gsum, idx, Wf1, bf1, Wf2, bf2,
                                                  gamma, beta, out);
}
